// KNeighborsVC_6545530159792
// MI455X (gfx1250) — compile-verified, matrix operand order checked by reading, not run on hardware
//
#include <hip/hip_runtime.h>
#include <math.h>

// Problem constants (from reference)
#define K_DIM   1024
#define NQ      2048
#define NC      100000
#define TOPK    4

// Tiling
#define CSPLIT  16            // candidate slices (grid.y)
#define SLICE   6250          // NC / CSPLIT exactly
#define QT      32            // queries per block (2 WMMA M-tiles)
#define CT      256           // candidates per chunk (8 waves x 2 N-tiles x 16)
#define NCHUNK  25            // ceil(SLICE / CT)
#define QROW    1032          // q_lds row stride in bf16 (1024 + 8 pad -> conflict-free b128)
#define SROW    36            // score_lds row stride in floats (32 + 4 pad)

typedef __attribute__((ext_vector_type(16))) __bf16        v16bf;
typedef __attribute__((ext_vector_type(8)))  float         v8f;
typedef __attribute__((ext_vector_type(4)))  unsigned int  u32x4;
typedef __attribute__((ext_vector_type(4)))  float         f32x4;
typedef __attribute__((ext_vector_type(4)))  __bf16        bf16x4;

union Frag16 { v16bf v; u32x4 u[2]; };
union AccU   { v8f   f; u32x4 u[2]; };

// ---------------------------------------------------------------------------
// Kernel 1: row-normalize matching set, convert to bf16. One block per row.
// ---------------------------------------------------------------------------
__global__ __launch_bounds__(256) void normalize_bf16_kernel(
    const float* __restrict__ in, __bf16* __restrict__ out) {
  const int row = blockIdx.x;
  const int t   = threadIdx.x;
  const f32x4 v = *(const f32x4*)(in + (size_t)row * K_DIM + t * 4);
  float s = v[0]*v[0] + v[1]*v[1] + v[2]*v[2] + v[3]*v[3];
  __shared__ float red[256];
  red[t] = s;
  __syncthreads();
  for (int o = 128; o > 0; o >>= 1) {
    if (t < o) red[t] += red[t + o];
    __syncthreads();
  }
  const float rn = rsqrtf(red[0]);
  bf16x4 o4;
  o4[0] = (__bf16)(v[0] * rn);
  o4[1] = (__bf16)(v[1] * rn);
  o4[2] = (__bf16)(v[2] * rn);
  o4[3] = (__bf16)(v[3] * rn);
  *(bf16x4*)(out + (size_t)row * K_DIM + t * 4) = o4;
}

// ---------------------------------------------------------------------------
// Kernel 2: plain f32 -> bf16 convert for the query (no normalization needed:
// per-query scaling does not change top-k ranking).
// ---------------------------------------------------------------------------
__global__ __launch_bounds__(256) void convert_bf16_kernel(
    const float* __restrict__ in, __bf16* __restrict__ out) {
  const size_t i = ((size_t)blockIdx.x * 256 + threadIdx.x) * 4;
  const f32x4 v = *(const f32x4*)(in + i);
  bf16x4 o4;
  o4[0] = (__bf16)v[0];
  o4[1] = (__bf16)v[1];
  o4[2] = (__bf16)v[2];
  o4[3] = (__bf16)v[3];
  *(bf16x4*)(out + i) = o4;
}

// Sorted-descending top-4 insert
#define TOP4_INSERT(v, gi)                                                     \
  if ((v) > tv3) {                                                             \
    if ((v) > tv1) {                                                           \
      if ((v) > tv0) { tv3=tv2; ti3=ti2; tv2=tv1; ti2=ti1; tv1=tv0; ti1=ti0;   \
                       tv0=(v); ti0=(gi); }                                    \
      else           { tv3=tv2; ti3=ti2; tv2=tv1; ti2=ti1; tv1=(v); ti1=(gi); }\
    } else {                                                                   \
      if ((v) > tv2) { tv3=tv2; ti3=ti2; tv2=(v); ti2=(gi); }                  \
      else           { tv3=(v); ti3=(gi); }                                    \
    }                                                                          \
  }

// ---------------------------------------------------------------------------
// Kernel 3: main fused score + streaming top-4.
// grid = (NQ/QT, CSPLIT), block = 256 (8 waves).
// Each wave: 2 M-tiles x 2 N-tiles of v_wmma_f32_16x16x32_bf16 over K=1024.
// ---------------------------------------------------------------------------
__global__ __launch_bounds__(256) void knn_score_topk_kernel(
    const __bf16* __restrict__ Qb, const __bf16* __restrict__ Mb,
    float* __restrict__ pv, int* __restrict__ pi) {
  __shared__ __bf16 q_lds[QT * QROW];
  __shared__ float  score_lds[CT * SROW];
  __shared__ float  mv[QT][8][4];
  __shared__ int    mi[QT][8][4];

  const int t     = threadIdx.x;
  const int lane  = t & 31;
  const int wave  = t >> 5;
  const int l16   = lane & 15;
  const int half  = lane >> 4;            // 0: K+0 block, 1: K+8 block
  const int qbase = blockIdx.x * QT;
  const int slice_start = blockIdx.y * SLICE;
  const int slice_end   = slice_start + SLICE;   // CSPLIT*SLICE == NC exactly

  // Stage query tile into LDS (32 rows x 1024 bf16, padded rows).
  for (int idx = t; idx < QT * (K_DIM / 8); idx += 256) {
    const int row  = idx >> 7;            // / 128
    const int col8 = idx & 127;
    *(u32x4*)&q_lds[row * QROW + col8 * 8] =
        *(const u32x4*)(Qb + (size_t)(qbase + row) * K_DIM + col8 * 8);
  }
  __syncthreads();

  float tv0 = -INFINITY, tv1 = -INFINITY, tv2 = -INFINITY, tv3 = -INFINITY;
  int   ti0 = 0, ti1 = 0, ti2 = 0, ti3 = 0;

  const __bf16* pA0 = q_lds + (size_t)l16 * QROW;          // M-tile 0
  const __bf16* pA1 = q_lds + (size_t)(16 + l16) * QROW;   // M-tile 1
  const int kh = half * 8;

  v8f NEGINF8 = { -INFINITY, -INFINITY, -INFINITY, -INFINITY,
                  -INFINITY, -INFINITY, -INFINITY, -INFINITY };

  for (int chunk = 0; chunk < NCHUNK; ++chunk) {
    const int cchunk = slice_start + chunk * CT;
    const int cloc   = wave * 32;                 // this wave's N-strip base
    const int candA  = cchunk + cloc + l16;       // N-tile 0
    const int candB  = candA + 16;                // N-tile 1
    const bool vA = candA < slice_end;
    const bool vB = candB < slice_end;
    const __bf16* pB0 = Mb + (size_t)(vA ? candA : slice_start) * K_DIM;
    const __bf16* pB1 = Mb + (size_t)(vB ? candB : slice_start) * K_DIM;

    v8f acc00 = {}, acc01 = {}, acc10 = {}, acc11 = {};

#pragma unroll 4
    for (int kc = 0; kc < K_DIM; kc += 32) {
      const int ko = kc + kh;
      Frag16 a0, a1, b0, b1;
      a0.u[0] = *(const u32x4*)(pA0 + ko);
      a0.u[1] = *(const u32x4*)(pA0 + ko + 16);
      a1.u[0] = *(const u32x4*)(pA1 + ko);
      a1.u[1] = *(const u32x4*)(pA1 + ko + 16);
      b0.u[0] = *(const u32x4*)(pB0 + ko);
      b0.u[1] = *(const u32x4*)(pB0 + ko + 16);
      b1.u[0] = *(const u32x4*)(pB1 + ko);
      b1.u[1] = *(const u32x4*)(pB1 + ko + 16);
      acc00 = __builtin_amdgcn_wmma_f32_16x16x32_bf16(false, a0.v, false, b0.v,
                                                      (short)0, acc00, false, false);
      acc01 = __builtin_amdgcn_wmma_f32_16x16x32_bf16(false, a0.v, false, b1.v,
                                                      (short)0, acc01, false, false);
      acc10 = __builtin_amdgcn_wmma_f32_16x16x32_bf16(false, a1.v, false, b0.v,
                                                      (short)0, acc10, false, false);
      acc11 = __builtin_amdgcn_wmma_f32_16x16x32_bf16(false, a1.v, false, b1.v,
                                                      (short)0, acc11, false, false);
    }

    // Store scores to LDS: score_lds[c_local * SROW + q_local]; mask invalid
    // candidates with -inf so the top-4 scan ignores them.
    {
      AccU s;
      float* d0 = &score_lds[(cloc + l16) * SROW + half * 8];        // nt=0
      float* d1 = &score_lds[(cloc + 16 + l16) * SROW + half * 8];   // nt=1
      s.f = vA ? acc00 : NEGINF8;                   // mt=0, q 0..15
      *(u32x4*)(d0)          = s.u[0];
      *(u32x4*)(d0 + 4)      = s.u[1];
      s.f = vA ? acc10 : NEGINF8;                   // mt=1, q 16..31
      *(u32x4*)(d0 + 16)     = s.u[0];
      *(u32x4*)(d0 + 20)     = s.u[1];
      s.f = vB ? acc01 : NEGINF8;
      *(u32x4*)(d1)          = s.u[0];
      *(u32x4*)(d1 + 4)      = s.u[1];
      s.f = vB ? acc11 : NEGINF8;
      *(u32x4*)(d1 + 16)     = s.u[0];
      *(u32x4*)(d1 + 20)     = s.u[1];
    }
    __syncthreads();

    // Streaming per-thread top-4: thread owns query (t&31) over 32 candidates.
    {
      const int q     = t & 31;
      const int strip = t >> 5;
      const int cb    = strip * 32;
      for (int j = 0; j < 32; ++j) {
        const float v  = score_lds[(cb + j) * SROW + q];
        const int   gi = cchunk + cb + j;
        TOP4_INSERT(v, gi)
      }
    }
    __syncthreads();
  }

  // Merge the 8 per-strip partial top-4s for each query; emit slice top-4.
  {
    const int q     = t & 31;
    const int strip = t >> 5;
    mv[q][strip][0] = tv0; mv[q][strip][1] = tv1;
    mv[q][strip][2] = tv2; mv[q][strip][3] = tv3;
    mi[q][strip][0] = ti0; mi[q][strip][1] = ti1;
    mi[q][strip][2] = ti2; mi[q][strip][3] = ti3;
  }
  __syncthreads();
  if (t < QT) {
    const int q = t;
    float tv0 = -INFINITY, tv1 = -INFINITY, tv2 = -INFINITY, tv3 = -INFINITY;
    int   ti0 = 0, ti1 = 0, ti2 = 0, ti3 = 0;
    for (int s = 0; s < 8; ++s)
      for (int j = 0; j < 4; ++j) {
        const float v  = mv[q][s][j];
        const int   gi = mi[q][s][j];
        TOP4_INSERT(v, gi)
      }
    const int base = ((qbase + q) * CSPLIT + (int)blockIdx.y) * 4;
    pv[base + 0] = tv0; pv[base + 1] = tv1;
    pv[base + 2] = tv2; pv[base + 3] = tv3;
    pi[base + 0] = ti0; pi[base + 1] = ti1;
    pi[base + 2] = ti2; pi[base + 3] = ti3;
  }
}

// ---------------------------------------------------------------------------
// Kernel 4: merge CSPLIT slice top-4s -> global top-4, gather + average synth
// rows. One block per query.
// ---------------------------------------------------------------------------
__global__ __launch_bounds__(256) void merge_gather_kernel(
    const float* __restrict__ pv, const int* __restrict__ pi,
    const float* __restrict__ synth, float* __restrict__ out) {
  const int q = blockIdx.x;
  __shared__ int sidx[TOPK];
  if (threadIdx.x == 0) {
    float tv0 = -INFINITY, tv1 = -INFINITY, tv2 = -INFINITY, tv3 = -INFINITY;
    int   ti0 = 0, ti1 = 0, ti2 = 0, ti3 = 0;
    const float* vv = pv + (size_t)q * CSPLIT * 4;
    const int*   ii = pi + (size_t)q * CSPLIT * 4;
    for (int j = 0; j < CSPLIT * 4; ++j) {
      const float v  = vv[j];
      const int   gi = ii[j];
      TOP4_INSERT(v, gi)
    }
    sidx[0] = ti0; sidx[1] = ti1; sidx[2] = ti2; sidx[3] = ti3;
  }
  __syncthreads();
  const int fo = threadIdx.x * 4;
  const f32x4 a = *(const f32x4*)(synth + (size_t)sidx[0] * K_DIM + fo);
  const f32x4 b = *(const f32x4*)(synth + (size_t)sidx[1] * K_DIM + fo);
  const f32x4 c = *(const f32x4*)(synth + (size_t)sidx[2] * K_DIM + fo);
  const f32x4 d = *(const f32x4*)(synth + (size_t)sidx[3] * K_DIM + fo);
  const f32x4 r = (a + b + c + d) * 0.25f;
  *(f32x4*)(out + (size_t)q * K_DIM + fo) = r;
}

// ---------------------------------------------------------------------------
extern "C" void kernel_launch(void* const* d_in, const int* in_sizes, int n_in,
                              void* d_out, int out_size, void* d_ws, size_t ws_size,
                              hipStream_t stream) {
  (void)in_sizes; (void)n_in; (void)out_size; (void)ws_size;
  const float* query    = (const float*)d_in[0];   // (2048, 1024) f32
  const float* matching = (const float*)d_in[1];   // (100000, 1024) f32
  const float* synth    = (const float*)d_in[2];   // (100000, 1024) f32
  // d_in[3] = topk (==4), hardcoded

  char* ws = (char*)d_ws;
  __bf16* Mb = (__bf16*)ws;                                   // 204,800,000 B
  const size_t offQ  = (size_t)NC * K_DIM * 2;
  __bf16* Qb = (__bf16*)(ws + offQ);                          //   4,194,304 B
  const size_t offPV = offQ + (size_t)NQ * K_DIM * 2;
  float* pv = (float*)(ws + offPV);                           //     524,288 B
  const size_t offPI = offPV + (size_t)NQ * CSPLIT * 4 * sizeof(float);
  int* pi = (int*)(ws + offPI);                               //     524,288 B

  normalize_bf16_kernel<<<NC, 256, 0, stream>>>(matching, Mb);
  convert_bf16_kernel<<<(NQ * K_DIM) / (256 * 4), 256, 0, stream>>>(query, Qb);

  dim3 grid(NQ / QT, CSPLIT);
  knn_score_topk_kernel<<<grid, 256, 0, stream>>>(Qb, Mb, pv, pi);

  merge_gather_kernel<<<NQ, 256, 0, stream>>>(pv, pi, synth, (float*)d_out);
}
